// GSAGE_Close_52269751992820
// MI455X (gfx1250) — matrix operand order repaired and checked by reading, not compiled, not run on hardware
//
#include <hip/hip_runtime.h>
#include <math.h>

// ---------------------------------------------------------------------------
// GNN-Bet forward on MI455X (gfx1250): bf16 WMMA GEMMs with pre-transposed
// bf16 weight cache, row-stripe-per-wave GEMM (A fragment reused across all
// column tiles), fused MLP output reduction, prefetched atomic edge scatter.
// Wave32 everywhere.
// ---------------------------------------------------------------------------

typedef __attribute__((ext_vector_type(16))) __bf16 v16bf;
typedef __attribute__((ext_vector_type(8)))  float  v8f;

#define NHID 96

__device__ __forceinline__ float wave_reduce_sum(float v) {
#pragma unroll
    for (int m = 16; m >= 1; m >>= 1) v += __shfl_xor(v, m, 32);
    return v;
}

// ---------------- generic zero fill ----------------
__global__ void __launch_bounds__(256) fill_zero_f32(float* __restrict__ p, size_t n) {
    size_t i = (size_t)blockIdx.x * 256 + threadIdx.x;
    if (i < n) p[i] = 0.0f;
}

// ---------------- weight cache: Wt[b][n][k] = bf16(W[b][k][n]) ---------------
__global__ void __launch_bounds__(256)
convert_transpose_bf16(const float* __restrict__ W, __bf16* __restrict__ Wt,
                       int K, int ncols, int batch)
{
    size_t idx = (size_t)blockIdx.x * 256 + threadIdx.x;
    size_t per = (size_t)K * ncols;
    if (idx >= per * (size_t)batch) return;
    size_t b = idx / per;
    size_t r = idx - b * per;
    int k  = (int)(r / ncols);
    int nn = (int)(r - (size_t)k * ncols);
    Wt[b * per + (size_t)nn * K + k] = (__bf16)W[idx];
}

// ---------------- edge scatter: out[dst[e]] += X[src[e]] * val[e], F=96 ------
// one wave per 4 edges, 3 features per lane, prefetch upcoming gather rows
__global__ void __launch_bounds__(256)
scatter_edges_f96(const float* __restrict__ Xsrc, const float* __restrict__ vals,
                  const int* __restrict__ src, const int* __restrict__ dst,
                  float* __restrict__ out, int nE)
{
    int wave  = blockIdx.x * 8 + (threadIdx.x >> 5);
    int ebase = wave * 4;
    if (ebase >= nE) return;
    int lane = threadIdx.x & 31;

    int ss[4], dd[4]; float vv[4];
#pragma unroll
    for (int i = 0; i < 4; ++i) {
        int e = ebase + i;
        if (e < nE) { ss[i] = src[e]; dd[i] = dst[e]; vv[i] = vals ? vals[e] : 1.0f; }
        else        { ss[i] = ss[0];  dd[i] = -1;     vv[i] = 0.0f; }
    }
    // prefetch the gathered feature rows (96 f32 = 3 x 128B lines each)
#pragma unroll
    for (int i = 1; i < 4; ++i)
        if (lane < 3) __builtin_prefetch(Xsrc + (size_t)ss[i] * NHID + lane * 32, 0, 0);

#pragma unroll
    for (int i = 0; i < 4; ++i) {
        if (dd[i] < 0) continue;                   // uniform per wave
        const float* xp = Xsrc + (size_t)ss[i] * NHID;
        float*       op = out  + (size_t)dd[i] * NHID;
#pragma unroll
        for (int j = 0; j < 3; ++j) {
            int f = lane + 32 * j;
            atomicAdd(op + f, xp[f] * vv[i]);
        }
    }
}

// ---------------- degree count ----------------
__global__ void __launch_bounds__(256)
deg_count(const int* __restrict__ dst, float* __restrict__ deg, int nE) {
    int e = blockIdx.x * 256 + threadIdx.x;
    if (e < nE) atomicAdd(&deg[dst[e]], 1.0f);
}

// ---------------- per-row bias + relu + optional l2norm (F=96) ---------------
// one wave per node; safe with X == Y
__global__ void __launch_bounds__(256)
row_bias_relu_norm96(const float* __restrict__ X, const float* __restrict__ bias,
                     float* __restrict__ Y, int n, int do_norm)
{
    int node = blockIdx.x * 8 + (threadIdx.x >> 5);
    if (node >= n) return;
    int lane = threadIdx.x & 31;
    const float* xp = X + (size_t)node * NHID;
    float v[3];
#pragma unroll
    for (int j = 0; j < 3; ++j) {
        int f = lane + 32 * j;
        float t = xp[f] + (bias ? bias[f] : 0.0f);
        v[j] = fmaxf(t, 0.0f);
    }
    float inv = 1.0f;
    if (do_norm) {
        float ss = v[0]*v[0] + v[1]*v[1] + v[2]*v[2];
        ss = wave_reduce_sum(ss);
        inv = 1.0f / fmaxf(sqrtf(ss), 1e-12f);
    }
    float* yp = Y + (size_t)node * NHID;
#pragma unroll
    for (int j = 0; j < 3; ++j) yp[lane + 32 * j] = v[j] * inv;
}

// ---------------- WMMA GEMM: one wave owns a 16-row x (NCT*16)-col stripe ----
// Y = act( scaleA(Xa)@Wa [+ Xb@Wb] + bias )           (W3 == nullptr)
// score[row] += act(...) . W3 + b3                    (W3 != nullptr, fused dot)
// Weights pre-transposed bf16: WaT[n][k] -> lane's 16 B elems are contiguous.
// A fragment loaded/converted once per k-tile, reused for all NCT col tiles.
template<int NCT>
__global__ void __launch_bounds__(128)
gemm_bf16_wmma(const float* __restrict__ Xa, const float* __restrict__ deg,
               const __bf16* __restrict__ WaT,
               const float* __restrict__ Xb, const __bf16* __restrict__ WbT,
               const float* __restrict__ bias, float* __restrict__ Y,
               const float* __restrict__ W3, const float* __restrict__ b3,
               float* __restrict__ score,
               int nrows, int K, int do_relu)
{
    const int ncols = NCT * 16;
    const int lane = threadIdx.x;          // 0..31
    const int hi   = lane >> 4;            // 0 or 1
    const int ln   = lane & 15;
    const int rowTile = blockIdx.x * blockDim.y + threadIdx.y;
    if (rowTile * 16 >= nrows) return;     // uniform per wave -> EXEC stays full
    const int rowA = rowTile * 16 + ln;    // A row for this lane

    float scaleA = 1.0f;
    if (deg) scaleA = 1.0f / fmaxf(deg[rowA], 1.0f);

    v8f c[NCT];
#pragma unroll
    for (int ct = 0; ct < NCT; ++ct) c[ct] = (v8f){};

    const int nkt = K >> 5;                // K multiple of 32 (96 or 192)
    for (int kt = 0; kt < nkt; ++kt) {
        const int kbase = kt * 32;
        // A fragments (loaded once per k-tile, reused across all col tiles)
        v16bf aa, ab;
        {
            const float* ap = Xa + (size_t)rowA * K + kbase + hi * 16;
#pragma unroll
            for (int j = 0; j < 8; ++j) aa[j]     = (__bf16)(ap[j]      * scaleA);
#pragma unroll
            for (int j = 0; j < 8; ++j) aa[8 + j] = (__bf16)(ap[8 + j] * scaleA);
        }
        if (Xb) {
            const float* ap = Xb + (size_t)rowA * K + kbase + hi * 16;
#pragma unroll
            for (int j = 0; j < 8; ++j) ab[j]     = (__bf16)ap[j];
#pragma unroll
            for (int j = 0; j < 8; ++j) ab[8 + j] = (__bf16)ap[8 + j];
        }
        // B: lane ln covers column ct*16+ln; 16 contiguous bf16 (32B) per tile
        const __bf16* wa = WaT + (size_t)ln * K + kbase + hi * 16;
#pragma unroll
        for (int ct = 0; ct < NCT; ++ct) {
            v16bf b = *(const v16bf*)(wa + (size_t)ct * 16 * K);
            c[ct] = __builtin_amdgcn_wmma_f32_16x16x32_bf16(
                        false, aa, false, b, (short)0, c[ct], false, false);
        }
        if (Xb) {
            const __bf16* wb = WbT + (size_t)ln * K + kbase + hi * 16;
#pragma unroll
            for (int ct = 0; ct < NCT; ++ct) {
                v16bf b = *(const v16bf*)(wb + (size_t)ct * 16 * K);
                c[ct] = __builtin_amdgcn_wmma_f32_16x16x32_bf16(
                            false, ab, false, b, (short)0, c[ct], false, false);
            }
        }
    }

    if (W3) {
        // whole row dot lives in this wave: no atomics, b3 added exactly once
        float part[8];
#pragma unroll
        for (int v = 0; v < 8; ++v) part[v] = 0.0f;
#pragma unroll
        for (int ct = 0; ct < NCT; ++ct) {
            const int col = ct * 16 + ln;
            const float bv  = bias ? bias[col] : 0.0f;
            const float w3v = W3[col];
#pragma unroll
            for (int v = 0; v < 8; ++v) {
                float o = c[ct][v] + bv;
                if (do_relu) o = fmaxf(o, 0.0f);
                part[v] += o * w3v;
            }
        }
#pragma unroll
        for (int m = 8; m >= 1; m >>= 1) {
#pragma unroll
            for (int v = 0; v < 8; ++v) part[v] += __shfl_xor(part[v], m, 32);
        }
        if (ln == 0) {
#pragma unroll
            for (int v = 0; v < 8; ++v) {
                int row = rowTile * 16 + v + hi * 8;
                score[row] += part[v] + b3[0];
            }
        }
    } else {
#pragma unroll
        for (int ct = 0; ct < NCT; ++ct) {
            const int col = ct * 16 + ln;
            const float bv = bias ? bias[col] : 0.0f;
#pragma unroll
            for (int v = 0; v < 8; ++v) {
                int row = rowTile * 16 + v + hi * 8;   // C: VGPR v -> M=v+hi*8
                float o = c[ct][v] + bv;
                if (do_relu) o = fmaxf(o, 0.0f);
                Y[(size_t)row * ncols + col] = o;
            }
        }
    }
}

// ---------------------------------------------------------------------------
extern "C" void kernel_launch(void* const* d_in, const int* in_sizes, int n_in,
                              void* d_out, int out_size, void* d_ws, size_t ws_size,
                              hipStream_t stream)
{
    const int*   ei1     = (const int*)  d_in[0];
    const float* vals1   = (const float*)d_in[1];
    const int*   ei2     = (const int*)  d_in[2];
    const float* gc1_W   = (const float*)d_in[3];
    const float* gc1_b   = (const float*)d_in[4];
    const float* sage_Wl = (const float*)d_in[5];
    const float* sage_bl = (const float*)d_in[6];
    const float* sage_Wr = (const float*)d_in[7];
    const float* W1      = (const float*)d_in[8];
    const float* b1      = (const float*)d_in[9];
    const float* W2      = (const float*)d_in[10];
    const float* b2      = (const float*)d_in[11];
    const float* W3      = (const float*)d_in[12];
    const float* b3      = (const float*)d_in[13];

    const int E1 = in_sizes[0] / 2;
    const int E2 = in_sizes[2] / 2;
    const int n  = in_sizes[3] / NHID;          // 50000

    const int* dst1 = ei1;        // edge_index1[0]
    const int* src1 = ei1 + E1;   // edge_index1[1]
    const int* src2 = ei2;        // edge_index2[0]
    const int* dst2 = ei2 + E2;   // edge_index2[1]

    float* ws = (float*)d_ws;
    const size_t NF = (size_t)n * NHID;
    float* x21  = ws;                                  // N*96
    float* agg0 = ws + 1 * NF;                         // N*96 (segsum of x21)
    float* bufA = ws + 2 * NF;                         // N*96
    float* bufB = ws + 3 * NF;                         // N*96
    float* aggT = ws + 4 * NF;                         // N*96
    float* h1   = ws + 5 * NF;                         // N*192
    float* deg  = h1 + (size_t)n * 192;                // N
    // bf16 transposed weight cache (32B-aligned by construction)
    __bf16* wlt = (__bf16*)(deg + n);                  // 7 x [96][96]
    __bf16* wrt = wlt + (size_t)7 * NHID * NHID;       // 7 x [96][96]
    __bf16* w1t = wrt + (size_t)7 * NHID * NHID;       // [192][96]
    __bf16* w2t = w1t + (size_t)NHID * 192;            // [192][192]
    float* score = (float*)d_out;                      // [N,1] f32

    auto zero = [&](float* p, size_t cnt) {
        fill_zero_f32<<<(unsigned)((cnt + 255) / 256), 256, 0, stream>>>(p, cnt);
    };
    auto convert = [&](const float* W, __bf16* Wt, int K, int ncols, int batch) {
        size_t tot = (size_t)K * ncols * batch;
        convert_transpose_bf16<<<(unsigned)((tot + 255) / 256), 256, 0, stream>>>(
            W, Wt, K, ncols, batch);
    };
    auto gemm = [&](const float* Xa, const float* degp, const __bf16* Wa,
                    const float* Xb, const __bf16* Wb, const float* bias,
                    float* Y, int K, int ncols, int relu,
                    const float* W3p, const float* b3p, float* sc) {
        dim3 blk(32, 4);
        dim3 grd((n / 16 + 3) / 4);
        if (ncols == 96)
            gemm_bf16_wmma<6><<<grd, blk, 0, stream>>>(Xa, degp, Wa, Xb, Wb, bias,
                                                       Y, W3p, b3p, sc, n, K, relu);
        else
            gemm_bf16_wmma<12><<<grd, blk, 0, stream>>>(Xa, degp, Wa, Xb, Wb, bias,
                                                        Y, W3p, b3p, sc, n, K, relu);
    };
    auto mlp = [&](const float* x) {
        gemm(x,  nullptr, w1t, nullptr, nullptr, b1, h1, NHID, 192, 1,
             nullptr, nullptr, nullptr);
        gemm(h1, nullptr, w2t, nullptr, nullptr, b2, nullptr, 192, 192, 1,
             W3, b3, score);                       // fused dot into score
    };
    const unsigned gE32  = (unsigned)((E2 + 31) / 32);   // 8 waves x 4 edges
    const unsigned gNode = (unsigned)((n + 7) / 8);

    // ---- build bf16 transposed weight cache ----
    convert(sage_Wl, wlt, NHID, NHID, 7);
    convert(sage_Wr, wrt, NHID, NHID, 7);
    convert(W1,      w1t, NHID, 192,  1);
    convert(W2,      w2t, 192,  192,  1);

    // ---- gc1: h0 = segsum(gc1_W[src1]*val, dst1) + b; x2_1 = l2norm(relu(h0))
    zero(bufA, NF);
    scatter_edges_f96<<<(unsigned)((E1 + 31) / 32), 256, 0, stream>>>(
        gc1_W, vals1, src1, dst1, bufA, E1);
    row_bias_relu_norm96<<<gNode, 256, 0, stream>>>(bufA, gc1_b, x21, n, 1);

    // ---- degree of graph2 ----
    zero(deg, (size_t)n);
    deg_count<<<(unsigned)((E2 + 255) / 256), 256, 0, stream>>>(dst2, deg, E2);

    // ---- agg0 = segsum(x21[src2], dst2)  (trunk layer 0 + all 6 branches) ----
    zero(agg0, NF);
    scatter_edges_f96<<<gE32, 256, 0, stream>>>(x21, nullptr, src2, dst2, agg0, E2);

    // ---- score accumulator ----
    zero(score, (size_t)n);

    // ---- trunk: 6 intermediate SAGE layers ----
    const float* x = x21;
    float* ping = bufA;
    float* pong = bufB;
    for (int i = 0; i < 6; ++i) {
        const __bf16* Wl = wlt + (size_t)i * NHID * NHID;
        const __bf16* Wr = wrt + (size_t)i * NHID * NHID;
        const float*  bl = sage_bl + (size_t)i * NHID;
        const float* aggx;
        if (i == 0) {
            aggx = agg0;
        } else {
            zero(aggT, NF);
            scatter_edges_f96<<<gE32, 256, 0, stream>>>(x, nullptr, src2, dst2, aggT, E2);
            aggx = aggT;
        }
        gemm(aggx, deg, Wl, x, Wr, bl, ping, NHID, NHID, 1,
             nullptr, nullptr, nullptr);                       // relu fused
        row_bias_relu_norm96<<<gNode, 256, 0, stream>>>(ping, nullptr, ping, n, 1);
        x = ping;
        float* t = ping; ping = pong; pong = t;
    }

    // ---- x_last = relu(sage(x6, layer 6)), no norm; x ends in bufB ----
    zero(aggT, NF);
    scatter_edges_f96<<<gE32, 256, 0, stream>>>(x, nullptr, src2, dst2, aggT, E2);
    float* xlast = ping;   // == bufA after 6 swaps
    gemm(aggT, deg, wlt + (size_t)6 * NHID * NHID, x,
         wrt + (size_t)6 * NHID * NHID, sage_bl + (size_t)6 * NHID,
         xlast, NHID, NHID, 1, nullptr, nullptr, nullptr);

    // ---- scores ----
    mlp(x21);
    float* xi = pong;      // == bufB, trunk x6 no longer needed
    for (int i = 0; i < 6; ++i) {   // branches reuse agg0 (identical input)
        const __bf16* Wl = wlt + (size_t)i * NHID * NHID;
        const __bf16* Wr = wrt + (size_t)i * NHID * NHID;
        const float*  bl = sage_bl + (size_t)i * NHID;
        gemm(agg0, deg, Wl, x21, Wr, bl, xi, NHID, NHID, 1,
             nullptr, nullptr, nullptr);
        row_bias_relu_norm96<<<gNode, 256, 0, stream>>>(xi, nullptr, xi, n, 1);
        mlp(xi);
    }
    mlp(xlast);
}
